// PointNetSetAbstraction_10213432230205
// MI455X (gfx1250) — matrix operand order repaired and checked by reading, not compiled, not run on hardware
//
// PointNet++ Set Abstraction for MI455X (gfx1250, wave32, WMMA).
// Round 2: A-fragment reuse across all column tiles (one 16-row strip per wave,
// 4 or 8 accumulators) so each activation byte is read from HBM exactly once;
// final kernel stages the activated group tile in LDS (ds-based A fragments).
// Requires ws_size >= ~119 MB.
#include <hip/hip_runtime.h>
#include <cstdint>
#include <cstddef>

typedef __attribute__((ext_vector_type(16))) _Float16 v16h;
typedef __attribute__((ext_vector_type(8)))  _Float16 v8h;
typedef __attribute__((ext_vector_type(8)))  float    v8f;

#define NB    8
#define NPTS  8192
#define ND    64
#define NPOINT 1024
#define NSAMP 32
#define MROWS (NB * NPOINT * NSAMP)   // 262144
#define KPAD  96                      // 67 padded to 3 WMMA K-steps
#define R2    0.04f
#define BN_EPS 1e-5f

// ---------------- WMMA helpers (CDNA5 16x16x32 f16 -> f32) ----------------

__device__ __forceinline__ v8f wmma16(v16h a, v16h b, v8f c) {
    // emits v_wmma_f32_16x16x32_f16
    return __builtin_amdgcn_wmma_f32_16x16x32_f16(
        /*neg_a=*/false, a, /*neg_b=*/false, b,
        /*c_mod=*/(short)0, c, /*reuse_a=*/false, /*reuse_b=*/false);
}

// A-matrix 16x32 f16 fragment from row-major f16 (ISA 7.12.2 layout):
// lanes 0-15 row M=lane, halves 0-7 = K 0..7, halves 8-15 = K 16..23;
// lanes 16-31 same rows, K ranges shifted by +8.
__device__ __forceinline__ v16h load_a_frag(const _Float16* base, int stride,
                                            int mBase, int kBase, int lane) {
    int row = mBase + (lane & 15);
    int hi  = (lane >> 4) & 1;
    const _Float16* p = base + (size_t)row * stride + kBase + hi * 8;
    v8h lo  = *(const v8h*)p;
    v8h hi8 = *(const v8h*)(p + 16);
    v16h a;
#pragma unroll
    for (int j = 0; j < 8; ++j) { a[j] = lo[j]; a[j + 8] = hi8[j]; }
    return a;
}

// Same, but applies folded BatchNorm+ReLU per input channel k: relu(z*sa[k]+sc[k])
__device__ __forceinline__ v16h load_a_frag_act(const _Float16* base, int stride,
                                                int mBase, int kBase, int lane,
                                                const float* sa, const float* sc) {
    int row = mBase + (lane & 15);
    int hi  = (lane >> 4) & 1;
    int k0  = kBase + hi * 8;
    const _Float16* p = base + (size_t)row * stride + k0;
    v8h lo  = *(const v8h*)p;
    v8h hi8 = *(const v8h*)(p + 16);
    v16h a;
#pragma unroll
    for (int j = 0; j < 8; ++j) {
        int ka = k0 + j, kb = k0 + 16 + j;
        a[j]     = (_Float16)fmaxf(0.f, (float)lo[j]  * sa[ka] + sc[ka]);
        a[j + 8] = (_Float16)fmaxf(0.f, (float)hi8[j] * sa[kb] + sc[kb]);
    }
    return a;
}

// C/D 16x16 f32: lane holds column (lane&15); VGPR v holds row v (+8 for hi lanes)
__device__ __forceinline__ void store_d_f16(_Float16* Z, int stride,
                                            int mBase, int nBase, int lane, v8f c) {
    int col = nBase + (lane & 15);
    int rb  = (lane >> 4) ? 8 : 0;
#pragma unroll
    for (int v = 0; v < 8; ++v)
        Z[(size_t)(mBase + rb + v) * stride + col] = (_Float16)c[v];
}

// per-column sum / sum-of-squares epilogue for BatchNorm stats
__device__ __forceinline__ void stats_epilogue(float* gsum, float* gsq,
                                               int nBase, int lane, v8f c) {
    float s = 0.f, q = 0.f;
#pragma unroll
    for (int v = 0; v < 8; ++v) { float x = c[v]; s += x; q += x * x; }
    s += __shfl_xor(s, 16, 32);
    q += __shfl_xor(q, 16, 32);
    if (lane < 16) {
        atomicAdd(&gsum[nBase + lane], s);
        atomicAdd(&gsq[nBase + lane], q);
    }
}

// ---------------- Stage 1: farthest point sampling ----------------
// One block per batch; each of 1024 threads owns 8 points in registers.
__global__ __launch_bounds__(1024) void fps_kernel(const float* __restrict__ xyz,
                                                   float* __restrict__ out_newxyz) {
    const int b = blockIdx.x, t = threadIdx.x;
    const float* X = xyz + (size_t)b * NPTS * 3;
    float px[8], py[8], pz[8], d[8];
#pragma unroll
    for (int i = 0; i < 8; ++i) {
        int p = t * 8 + i;
        px[i] = X[p * 3 + 0]; py[i] = X[p * 3 + 1]; pz[i] = X[p * 3 + 2];
        d[i] = 1e10f;
    }
    __shared__ float sC[3];
    __shared__ float sVal[32];
    __shared__ int   sIdx[32];
    __shared__ int   sFar;
    int far = 0;
    const int wave = t >> 5, lane = t & 31;
    for (int it = 0; it < NPOINT; ++it) {
        if (t == (far >> 3)) {        // owner exports centroid coords
            int i = far & 7;
            sC[0] = px[i]; sC[1] = py[i]; sC[2] = pz[i];
        }
        __syncthreads();
        float cx = sC[0], cy = sC[1], cz = sC[2];
        if (t == 0) {                 // idx[it] = far (pre-update, per reference scan)
            float* o = out_newxyz + ((size_t)b * NPOINT + it) * 3;
            o[0] = cx; o[1] = cy; o[2] = cz;
        }
        float bv = -1.f; int bi = 0;
#pragma unroll
        for (int i = 0; i < 8; ++i) {
            float dx = px[i] - cx, dy = py[i] - cy, dz = pz[i] - cz;
            float dd = dx * dx + dy * dy + dz * dz;
            d[i] = fminf(d[i], dd);
            if (d[i] > bv) { bv = d[i]; bi = t * 8 + i; }
        }
#pragma unroll
        for (int off = 16; off > 0; off >>= 1) {   // wave argmax (first-occurrence ties)
            float ov = __shfl_down(bv, off, 32);
            int   oi = __shfl_down(bi, off, 32);
            if (ov > bv || (ov == bv && oi < bi)) { bv = ov; bi = oi; }
        }
        if (lane == 0) { sVal[wave] = bv; sIdx[wave] = bi; }
        __syncthreads();
        if (wave == 0) {
            bv = sVal[lane]; bi = sIdx[lane];
#pragma unroll
            for (int off = 16; off > 0; off >>= 1) {
                float ov = __shfl_down(bv, off, 32);
                int   oi = __shfl_down(bi, off, 32);
                if (ov > bv || (ov == bv && oi < bi)) { bv = ov; bi = oi; }
            }
            if (lane == 0) sFar = bi;
        }
        __syncthreads();
        far = sFar;
    }
}

// ---------------- Stage 2: ball query (first 32 indices in ascending order) ----
__global__ __launch_bounds__(256) void ballq_kernel(const float* __restrict__ xyz,
                                                    const float* __restrict__ newxyz,
                                                    int* __restrict__ gidx) {
    const int wave = threadIdx.x >> 5, lane = threadIdx.x & 31;
    const int cid  = blockIdx.x * 8 + wave;     // 0..8191
    const int b    = cid >> 10;
    const float* X = xyz + (size_t)b * NPTS * 3;
    const float cx = newxyz[cid * 3 + 0];
    const float cy = newxyz[cid * 3 + 1];
    const float cz = newxyz[cid * 3 + 2];
    int* G = gidx + (size_t)cid * NSAMP;
    int cnt = 0, first = 0;
    for (int ch = 0; ch < NPTS / 32; ++ch) {
        int p = ch * 32 + lane;
        float dx = X[p * 3 + 0] - cx;
        float dy = X[p * 3 + 1] - cy;
        float dz = X[p * 3 + 2] - cz;
        bool cond = (dx * dx + dy * dy + dz * dz) <= R2;
        unsigned m = (unsigned)__ballot(cond);
        if (cond) {
            int slot = cnt + __popc(m & ((1u << lane) - 1u));
            if (slot < NSAMP) G[slot] = p;
        }
        if (cnt == 0 && m != 0) first = __shfl(p, __ffs(m) - 1, 32);
        cnt += __popc(m);
        if (cnt >= NSAMP) break;
    }
    for (int slot = cnt + lane; slot < NSAMP; slot += 32) G[slot] = first;
}

// ---------------- Stage 3: feature build (f16, K padded to 96) ----------------
__global__ __launch_bounds__(256) void build_feat_kernel(const float* __restrict__ xyz,
                                                         const float* __restrict__ points,
                                                         const int* __restrict__ gidx,
                                                         const float* __restrict__ newxyz,
                                                         _Float16* __restrict__ Xh) {
    const int total = MROWS * KPAD;
    for (int e = blockIdx.x * blockDim.x + threadIdx.x; e < total;
         e += gridDim.x * blockDim.x) {
        int ch = e % KPAD;
        int m  = e / KPAD;
        int n  = m & 31;
        int sg = m >> 5;          // b*1024 + s
        int b  = sg >> 10;
        int g  = gidx[(size_t)sg * NSAMP + n];
        float v;
        if (ch < 3)
            v = xyz[((size_t)b * NPTS + g) * 3 + ch] - newxyz[(size_t)sg * 3 + ch];
        else if (ch < 67)
            v = points[((size_t)b * NPTS + g) * ND + (ch - 3)];
        else
            v = 0.f;
        Xh[(size_t)m * KPAD + ch] = (_Float16)v;
    }
}

// ---------------- Weight pre-swizzle into WMMA B-fragment layout ----------------
// Wp[((kt*NT + nt)*32 + lane)*16 + j] with col = nt*16 + (lane&15),
// k = kt*32 + (lane>=16 ? 8 : 0) + (j<8 ? j : j+8). Zero-pad k >= Kin.
__global__ void prep_w_kernel(const float* __restrict__ W, _Float16* __restrict__ Wp,
                              int Kin, int KT, int NT) {
    int total = KT * NT * 512;
    for (int e = blockIdx.x * blockDim.x + threadIdx.x; e < total;
         e += gridDim.x * blockDim.x) {
        int j  = e & 15;
        int l  = (e >> 4) & 31;
        int r  = e >> 9;
        int nt = r % NT, kt = r / NT;
        int col = nt * 16 + (l & 15);
        int k   = kt * 32 + ((l >> 4) ? 8 : 0) + (j < 8 ? j : j + 8);
        Wp[e] = (_Float16)((k < Kin) ? W[(size_t)col * Kin + k] : 0.f);
    }
}

__global__ void zero_stats_kernel(float* stats) { stats[threadIdx.x] = 0.f; }

// mean/var -> folded affine: a = g * rsqrt(var+eps); c = b - mean*a
__global__ void finalize_kernel(const float* __restrict__ sum, const float* __restrict__ sq,
                                const float* __restrict__ gamma, const float* __restrict__ beta,
                                float* __restrict__ a, float* __restrict__ c, int Nc) {
    int t = threadIdx.x;
    if (t < Nc) {
        const float invM = 1.f / (float)MROWS;
        float mean = sum[t] * invM;
        float var  = sq[t] * invM - mean * mean;
        float s    = gamma[t] * rsqrtf(var + BN_EPS);
        a[t] = s;
        c[t] = beta[t] - mean * s;
    }
}

// ------- GEMM layer 1: Z1 = X(262144x96) * W1^T -> f16, + stats ----------
// One 16-row strip per wave, A loaded once per K-step, 4 accumulators (64 cols).
__global__ __launch_bounds__(256) void gemm1_kernel(const _Float16* __restrict__ Xh,
                                                    const _Float16* __restrict__ Wp1,
                                                    _Float16* __restrict__ Z1,
                                                    float* __restrict__ sum1,
                                                    float* __restrict__ sq1) {
    __shared__ alignas(32) _Float16 sW[3 * 4 * 512];   // 12 KB
    {
        const uint4* src = (const uint4*)Wp1;
        uint4* dst = (uint4*)sW;
        for (int i = threadIdx.x; i < 768; i += 256) dst[i] = src[i];
    }
    __syncthreads();
    const int wave = threadIdx.x >> 5, lane = threadIdx.x & 31;
    const int strip = blockIdx.x * 8 + wave;   // 16384 strips
    const int mBase = strip << 4;
    v8f acc[4] = {};
#pragma unroll
    for (int kt = 0; kt < 3; ++kt) {
        v16h a = load_a_frag(Xh, KPAD, mBase, kt * 32, lane);
#pragma unroll
        for (int nt = 0; nt < 4; ++nt) {
            v16h b = *(const v16h*)(sW + ((kt * 4 + nt) * 32 + lane) * 16);
            acc[nt] = wmma16(a, b, acc[nt]);
        }
    }
#pragma unroll
    for (int nt = 0; nt < 4; ++nt) {
        store_d_f16(Z1, 64, mBase, nt * 16, lane, acc[nt]);
        stats_epilogue(sum1, sq1, nt * 16, lane, acc[nt]);
    }
}

// ------- GEMM layer 2: Z2 = relu(BN1(Z1)) * W2^T, + stats ----------------
__global__ __launch_bounds__(256) void gemm2_kernel(const _Float16* __restrict__ Z1,
                                                    const _Float16* __restrict__ Wp2,
                                                    const float* __restrict__ stats,
                                                    _Float16* __restrict__ Z2,
                                                    float* __restrict__ sum2,
                                                    float* __restrict__ sq2) {
    __shared__ alignas(32) _Float16 sW[2 * 4 * 512];   // 8 KB
    __shared__ float sA[64], sC[64];
    {
        const uint4* src = (const uint4*)Wp2;
        uint4* dst = (uint4*)sW;
        for (int i = threadIdx.x; i < 512; i += 256) dst[i] = src[i];
        if (threadIdx.x < 64) {
            sA[threadIdx.x] = stats[128 + threadIdx.x];   // a1
            sC[threadIdx.x] = stats[192 + threadIdx.x];   // c1
        }
    }
    __syncthreads();
    const int wave = threadIdx.x >> 5, lane = threadIdx.x & 31;
    const int strip = blockIdx.x * 8 + wave;
    const int mBase = strip << 4;
    v8f acc[4] = {};
#pragma unroll
    for (int kt = 0; kt < 2; ++kt) {
        v16h a = load_a_frag_act(Z1, 64, mBase, kt * 32, lane, sA, sC);
#pragma unroll
        for (int nt = 0; nt < 4; ++nt) {
            v16h b = *(const v16h*)(sW + ((kt * 4 + nt) * 32 + lane) * 16);
            acc[nt] = wmma16(a, b, acc[nt]);
        }
    }
#pragma unroll
    for (int nt = 0; nt < 4; ++nt) {
        store_d_f16(Z2, 64, mBase, nt * 16, lane, acc[nt]);
        stats_epilogue(sum2, sq2, nt * 16, lane, acc[nt]);
    }
}

// ------- GEMM layer 3, stats-only (Z3 never materialized) ----------------
__global__ __launch_bounds__(256) void gemm3_stats_kernel(const _Float16* __restrict__ Z2,
                                                          const _Float16* __restrict__ Wp3,
                                                          const float* __restrict__ stats,
                                                          float* __restrict__ sum3,
                                                          float* __restrict__ sq3) {
    __shared__ alignas(32) _Float16 sW[2 * 8 * 512];   // 16 KB
    __shared__ float sA[64], sC[64];
    {
        const uint4* src = (const uint4*)Wp3;
        uint4* dst = (uint4*)sW;
        for (int i = threadIdx.x; i < 1024; i += 256) dst[i] = src[i];
        if (threadIdx.x < 64) {
            sA[threadIdx.x] = stats[384 + threadIdx.x];   // a2
            sC[threadIdx.x] = stats[448 + threadIdx.x];   // c2
        }
    }
    __syncthreads();
    const int wave = threadIdx.x >> 5, lane = threadIdx.x & 31;
    const int strip = blockIdx.x * 8 + wave;
    const int mBase = strip << 4;
    v8f acc[8] = {};
#pragma unroll
    for (int kt = 0; kt < 2; ++kt) {
        v16h a = load_a_frag_act(Z2, 64, mBase, kt * 32, lane, sA, sC);
#pragma unroll
        for (int nt = 0; nt < 8; ++nt) {
            v16h b = *(const v16h*)(sW + ((kt * 8 + nt) * 32 + lane) * 16);
            acc[nt] = wmma16(a, b, acc[nt]);
        }
    }
#pragma unroll
    for (int nt = 0; nt < 8; ++nt)
        stats_epilogue(sum3, sq3, nt * 16, lane, acc[nt]);
}

// ------- Final: stage activated group tile in LDS, recompute 32x128,
//         BN3+ReLU, max over 32 samples ----------------------------------
__global__ __launch_bounds__(256) void final_kernel(const _Float16* __restrict__ Z2,
                                                    const _Float16* __restrict__ Wp3,
                                                    const float* __restrict__ stats,
                                                    float* __restrict__ out) {
    __shared__ alignas(32) _Float16 sW[2 * 8 * 512];   // 16 KB
    __shared__ alignas(32) _Float16 sY[32 * 64];       // activated group tile, 4 KB
    __shared__ float sA2[64], sC2[64];
    const int tid = threadIdx.x;
    const int g   = blockIdx.x;      // group = b*1024 + s, rows g*32..g*32+31
    {
        const uint4* src = (const uint4*)Wp3;
        uint4* dst = (uint4*)sW;
        for (int i = tid; i < 1024; i += 256) dst[i] = src[i];
        if (tid < 64) {
            sA2[tid] = stats[384 + tid];
            sC2[tid] = stats[448 + tid];
        }
    }
    __syncthreads();
    // stage act2(Z2 group) into LDS once (32x64 f16)
    for (int i = tid; i < 32 * 64; i += 256) {
        int row = i >> 6, ch = i & 63;
        float z = (float)Z2[(size_t)(g * 32 + row) * 64 + ch];
        sY[i] = (_Float16)fmaxf(0.f, z * sA2[ch] + sC2[ch]);
    }
    __syncthreads();
    const int wave = tid >> 5, lane = tid & 31;
    const int nt = wave;            // 8 column tiles of 16 -> 128 channels
    v8f c0 = {}, c1 = {};
    const int arow = lane & 15;
    const int ahi  = (lane >> 4) & 1;
#pragma unroll
    for (int kt = 0; kt < 2; ++kt) {
        v16h b = *(const v16h*)(sW + ((kt * 8 + nt) * 32 + lane) * 16);
        int k0 = kt * 32 + ahi * 8;
        // A fragment for rows 0..15 of the staged tile
        {
            const _Float16* p = &sY[arow * 64 + k0];
            v8h lo = *(const v8h*)p;
            v8h hi = *(const v8h*)(p + 16);
            v16h a;
#pragma unroll
            for (int j = 0; j < 8; ++j) { a[j] = lo[j]; a[j + 8] = hi[j]; }
            c0 = wmma16(a, b, c0);
        }
        // A fragment for rows 16..31
        {
            const _Float16* p = &sY[(16 + arow) * 64 + k0];
            v8h lo = *(const v8h*)p;
            v8h hi = *(const v8h*)(p + 16);
            v16h a;
#pragma unroll
            for (int j = 0; j < 8; ++j) { a[j] = lo[j]; a[j + 8] = hi[j]; }
            c1 = wmma16(a, b, c1);
        }
    }
    const int col = nt * 16 + (lane & 15);
    const float a3 = stats[768 + col], b3 = stats[896 + col];
    float m = 0.f;   // relu output is >= 0
#pragma unroll
    for (int v = 0; v < 8; ++v) {
        m = fmaxf(m, fmaxf(0.f, c0[v] * a3 + b3));
        m = fmaxf(m, fmaxf(0.f, c1[v] * a3 + b3));
    }
    m = fmaxf(m, __shfl_xor(m, 16, 32));
    if (lane < 16)
        out[(size_t)NB * NPOINT * 3 + (size_t)g * 128 + col] = m;
}

// ---------------- host launcher ----------------

extern "C" void kernel_launch(void* const* d_in, const int* in_sizes, int n_in,
                              void* d_out, int out_size, void* d_ws, size_t ws_size,
                              hipStream_t stream) {
    const float* xyz    = (const float*)d_in[0];
    const float* points = (const float*)d_in[1];
    const float* W1 = (const float*)d_in[2];
    const float* g1 = (const float*)d_in[3];
    const float* b1 = (const float*)d_in[4];
    const float* W2 = (const float*)d_in[5];
    const float* g2 = (const float*)d_in[6];
    const float* b2 = (const float*)d_in[7];
    const float* W3 = (const float*)d_in[8];
    const float* g3 = (const float*)d_in[9];
    const float* b3 = (const float*)d_in[10];
    float* out = (float*)d_out;
    char*  ws  = (char*)d_ws;

    // workspace layout (bytes, 256-aligned)
    const size_t OFF_GIDX  = 0;                               // 8192*32*4 = 1 MB
    const size_t OFF_STATS = OFF_GIDX  + (size_t)8192 * 32 * 4;     // 1024 floats
    const size_t OFF_WP1   = OFF_STATS + 4096;                // 6144 halves
    const size_t OFF_WP2   = OFF_WP1   + 12288;               // 4096 halves
    const size_t OFF_WP3   = OFF_WP2   + 8192;                // 8192 halves
    const size_t OFF_XH    = OFF_WP3   + 16384;               // 262144*96 f16 = 48 MB
    const size_t OFF_Z1    = OFF_XH + (size_t)MROWS * KPAD * 2;   // 32 MB
    const size_t OFF_Z2    = OFF_Z1 + (size_t)MROWS * 64 * 2;     // 32 MB

    int*      gidx  = (int*)(ws + OFF_GIDX);
    float*    stats = (float*)(ws + OFF_STATS);
    _Float16* Wp1   = (_Float16*)(ws + OFF_WP1);
    _Float16* Wp2   = (_Float16*)(ws + OFF_WP2);
    _Float16* Wp3   = (_Float16*)(ws + OFF_WP3);
    _Float16* Xh    = (_Float16*)(ws + OFF_XH);
    _Float16* Z1    = (_Float16*)(ws + OFF_Z1);
    _Float16* Z2    = (_Float16*)(ws + OFF_Z2);

    // 1) FPS: writes new_xyz directly into d_out[0 .. 24576)
    fps_kernel<<<NB, 1024, 0, stream>>>(xyz, out);
    // 2) ball query (one wave per centroid)
    ballq_kernel<<<(NB * NPOINT) / 8, 256, 0, stream>>>(xyz, out, gidx);
    // 3) grouped feature tensor, f16, K padded to 96
    build_feat_kernel<<<32768, 256, 0, stream>>>(xyz, points, gidx, out, Xh);
    // 4) weights -> WMMA B-fragment layout
    prep_w_kernel<<<24, 256, 0, stream>>>(W1, Wp1, 67, 3, 4);
    prep_w_kernel<<<16, 256, 0, stream>>>(W2, Wp2, 64, 2, 4);
    prep_w_kernel<<<32, 256, 0, stream>>>(W3, Wp3, 64, 2, 8);
    zero_stats_kernel<<<1, 1024, 0, stream>>>(stats);
    // 5) layer 1 GEMM + stats, fold BN1   (16384 strips / 8 waves)
    gemm1_kernel<<<2048, 256, 0, stream>>>(Xh, Wp1, Z1, stats + 0, stats + 64);
    finalize_kernel<<<1, 64, 0, stream>>>(stats + 0, stats + 64, g1, b1,
                                          stats + 128, stats + 192, 64);
    // 6) layer 2 GEMM (BN1+ReLU folded into A-load) + stats, fold BN2
    gemm2_kernel<<<2048, 256, 0, stream>>>(Z1, Wp2, stats, Z2, stats + 256, stats + 320);
    finalize_kernel<<<1, 64, 0, stream>>>(stats + 256, stats + 320, g2, b2,
                                          stats + 384, stats + 448, 64);
    // 7) layer 3 stats-only GEMM pass, fold BN3
    gemm3_stats_kernel<<<2048, 256, 0, stream>>>(Z2, Wp3, stats, stats + 512, stats + 640);
    finalize_kernel<<<1, 128, 0, stream>>>(stats + 512, stats + 640, g3, b3,
                                           stats + 768, stats + 896, 128);
    // 8) recompute layer-3 tiles from LDS-staged activations, BN3+ReLU, max-pool
    final_kernel<<<NB * NPOINT, 256, 0, stream>>>(Z2, Wp3, stats, out);
}
